// CausalSelfAttention_1709396984226
// MI455X (gfx1250) — matrix operand order repaired and checked by reading, not compiled, not run on hardware
//
#include <hip/hip_runtime.h>
#include <hip/hip_bf16.h>

typedef __bf16 bf16_t;
typedef __attribute__((ext_vector_type(16))) __bf16 v16bf;
typedef __attribute__((ext_vector_type(8)))  __bf16 v8bf;
typedef __attribute__((ext_vector_type(8)))  float  v8f;

#define D_MODEL   1024
#define NHEAD     16
#define HEAD_DIM  64
#define SEQ       2048
#define BATCH     4
#define NROWS     (BATCH * SEQ)   // 8192

union AFrag { v16bf v; v8bf h[2]; };

// A-matrix 16x32 bf16 fragment from row-major [*, lda] at (row..row+15, k0..k0+31).
// ISA layout: lane<16 -> M=lane, K = e (e<8) / e+8 (e>=8); lane>=16 -> +8 on K.
__device__ __forceinline__ AFrag load_a_frag(const bf16_t* base, int lda, int row,
                                             int k0, int lane) {
  AFrag f;
  const int m    = lane & 15;
  const int koff = (lane >= 16) ? 8 : 0;
  const bf16_t* p = base + (size_t)(row + m) * lda + k0 + koff;
  f.h[0] = *(const v8bf*)(p);
  f.h[1] = *(const v8bf*)(p + 16);
  return f;
}

// B-matrix 32x16 bf16 fragment from a K-contiguous ("transposed", [N][ldb]) matrix.
// ISA layout: lane<16 -> N=lane, K = e (e<8) / e+8 (e>=8); lane>=16 -> +8 on K.
__device__ __forceinline__ v16bf load_b_frag(const bf16_t* baseT, int ldb, int col,
                                             int k0, int lane) {
  const int n     = lane & 15;
  const int kbase = (lane >= 16) ? 8 : 0;
  const bf16_t* p = baseT + (size_t)(col + n) * ldb + k0 + kbase;
  AFrag f;
  f.h[0] = *(const v8bf*)(p);
  f.h[1] = *(const v8bf*)(p + 16);
  return f.v;
}

__global__ void cvt_bf16_kernel(const float* __restrict__ in, bf16_t* __restrict__ out, int n) {
  int i = blockIdx.x * blockDim.x + threadIdx.x;
  if (i < n) out[i] = (bf16_t)in[i];
}

// in: [rows][cols] fp32  ->  out: [cols][rows] bf16
__global__ void transpose_cvt_kernel(const float* __restrict__ in, bf16_t* __restrict__ out,
                                     int rows, int cols) {
  int i = blockIdx.x * blockDim.x + threadIdx.x;
  if (i < rows * cols) {
    int r = i / cols, c = i % cols;
    out[(size_t)c * rows + r] = (bf16_t)in[i];
  }
}

// qkv = x @ W_qkv + b_qkv, scattered into per-head Q, K (row-major) and V^T buffers.
// One wave computes a 32x64 output tile: B fragments reused across two A fragments.
__global__ __launch_bounds__(32) void qkv_gemm_kernel(
    const bf16_t* __restrict__ A,     // [8192][1024] bf16
    const bf16_t* __restrict__ Bt,    // [3072][1024] bf16 (W_qkv^T)
    const float*  __restrict__ bias,  // [3072]
    bf16_t* __restrict__ qb,          // [B*H][S][64]
    bf16_t* __restrict__ kb,          // [B*H][S][64]
    bf16_t* __restrict__ vtb)         // [B*H][64][S]
{
  const int lane = threadIdx.x;
  const int mt = blockIdx.x;          // 256 tiles of 32 rows
  const int nt = blockIdx.y;          // 48 tiles of 64 cols
  v8f acc[2][4] = {};

  for (int k0 = 0; k0 < D_MODEL; k0 += 32) {
    AFrag af0 = load_a_frag(A, D_MODEL, mt * 32,      k0, lane);
    AFrag af1 = load_a_frag(A, D_MODEL, mt * 32 + 16, k0, lane);
#pragma unroll
    for (int j = 0; j < 4; ++j) {
      v16bf bfrag = load_b_frag(Bt, D_MODEL, nt * 64 + j * 16, k0, lane);
      acc[0][j] = __builtin_amdgcn_wmma_f32_16x16x32_bf16(false, af0.v, false, bfrag,
                                                          (short)0, acc[0][j], false, false);
      acc[1][j] = __builtin_amdgcn_wmma_f32_16x16x32_bf16(false, af1.v, false, bfrag,
                                                          (short)0, acc[1][j], false, false);
    }
  }

  const int n    = lane & 15;
  const int rowM = (lane >> 4) * 8;   // C layout: row = r + rowM
#pragma unroll
  for (int mi = 0; mi < 2; ++mi) {
#pragma unroll
    for (int j = 0; j < 4; ++j) {
      const int colg  = nt * 64 + j * 16 + n;   // [0, 3072)
      const float bv  = bias[colg];
      const int which = colg >> 10;             // 0=Q 1=K 2=V
      const int cq    = colg & 1023;
      const int h     = cq >> 6;
      const int d     = cq & 63;
#pragma unroll
      for (int r = 0; r < 8; ++r) {
        const int m = mt * 32 + mi * 16 + rowM + r;   // m = b*SEQ + s
        const int b = m >> 11;
        const int s = m & 2047;
        const bf16_t val = (bf16_t)(acc[mi][j][r] + bv);
        const size_t bh = (size_t)(b * NHEAD + h);
        if (which == 0)      qb [(bh * SEQ + s) * HEAD_DIM + d] = val;
        else if (which == 1) kb [(bh * SEQ + s) * HEAD_DIM + d] = val;
        else                 vtb[(bh * HEAD_DIM + d) * SEQ + s] = val;
      }
    }
  }
}

// Flash attention: one wave per (b, h, 16-row q tile). Online softmax in fp32.
__global__ __launch_bounds__(32) void attn_kernel(
    const bf16_t* __restrict__ qb,    // [B*H][S][64]
    const bf16_t* __restrict__ kb,    // [B*H][S][64]
    const bf16_t* __restrict__ vtb,   // [B*H][64][S]
    bf16_t* __restrict__ ob)          // [B*S][1024] bf16
{
  __shared__ __align__(32) bf16_t psh[16 * 32];   // P tile staging (C-layout -> A-layout)

  const int lane = threadIdx.x;
  const int qt = blockIdx.x;          // 128
  const int h  = blockIdx.y;          // 16
  const int b  = blockIdx.z;          // 4
  const size_t bh = (size_t)(b * NHEAD + h);
  const bf16_t* Q  = qb  + bh * SEQ * HEAD_DIM;
  const bf16_t* Km = kb  + bh * SEQ * HEAD_DIM;
  const bf16_t* Vt = vtb + bh * HEAD_DIM * SEQ;
  const int q0   = qt * 16;
  const int n    = lane & 15;
  const int rowM = (lane >> 4) * 8;

  AFrag qf[2];
#pragma unroll
  for (int c = 0; c < 2; ++c) qf[c] = load_a_frag(Q, HEAD_DIM, q0, c * 32, lane);

  float m_i[8], l_i[8];
  v8f acc[4] = {};
#pragma unroll
  for (int r = 0; r < 8; ++r) { m_i[r] = -__builtin_inff(); l_i[r] = 0.f; }

  const int npairs = (qt + 2) >> 1;   // ceil((qt+1)/2) 16-key tiles -> 32-key steps
  for (int p = 0; p < npairs; ++p) {
    const int k0 = p * 32;

    // scores: 16x32 tile as two 16x16 C fragments
    v8f sc[2];
#pragma unroll
    for (int j = 0; j < 2; ++j) {
      v8f c = {};
#pragma unroll
      for (int dc = 0; dc < 2; ++dc) {
        v16bf bfrag = load_b_frag(Km, HEAD_DIM, k0 + j * 16, dc * 32, lane);
        c = __builtin_amdgcn_wmma_f32_16x16x32_bf16(false, qf[dc].v, false, bfrag,
                                                    (short)0, c, false, false);
      }
      sc[j] = c;
    }

    // scale + causal mask + online softmax (row stats per lane half)
    const float scale = 0.125f;       // 1/sqrt(64)
    float alpha[8];
#pragma unroll
    for (int r = 0; r < 8; ++r) {
      const int rowg = q0 + rowM + r;
#pragma unroll
      for (int j = 0; j < 2; ++j) {
        const int colg = k0 + j * 16 + n;
        float v = (float)sc[j][r] * scale;
        v = (colg > rowg) ? -__builtin_inff() : v;
        sc[j][r] = v;
      }
      float rmax = fmaxf((float)sc[0][r], (float)sc[1][r]);
#pragma unroll
      for (int off = 8; off >= 1; off >>= 1)
        rmax = fmaxf(rmax, __shfl_xor(rmax, off, 16));
      const float mnew = fmaxf(m_i[r], rmax);
      alpha[r] = __expf(m_i[r] - mnew);
      float psum = 0.f;
#pragma unroll
      for (int j = 0; j < 2; ++j) {
        const float pv = __expf((float)sc[j][r] - mnew);
        sc[j][r] = pv;
        psum += pv;
      }
#pragma unroll
      for (int off = 8; off >= 1; off >>= 1)
        psum += __shfl_xor(psum, off, 16);
      l_i[r] = alpha[r] * l_i[r] + psum;
      m_i[r] = mnew;
    }

    // rescale accumulators
#pragma unroll
    for (int dt = 0; dt < 4; ++dt)
#pragma unroll
      for (int r = 0; r < 8; ++r)
        acc[dt][r] = acc[dt][r] * alpha[r];

    // P: C layout -> LDS (row-major 16x32) -> A fragment
    __syncthreads();
#pragma unroll
    for (int j = 0; j < 2; ++j)
#pragma unroll
      for (int r = 0; r < 8; ++r)
        psh[(rowM + r) * 32 + j * 16 + n] = (bf16_t)(float)sc[j][r];
    __syncthreads();

    AFrag pf;
    {
      const int koff = (lane >= 16) ? 8 : 0;
      const bf16_t* pp = psh + (lane & 15) * 32 + koff;
      pf.h[0] = *(const v8bf*)(pp);
      pf.h[1] = *(const v8bf*)(pp + 16);
    }

    // O += P @ V  (V^T gives K-contiguous B fragments)
#pragma unroll
    for (int dt = 0; dt < 4; ++dt) {
      v16bf vfrag = load_b_frag(Vt, SEQ, dt * 16, k0, lane);
      acc[dt] = __builtin_amdgcn_wmma_f32_16x16x32_bf16(false, pf.v, false, vfrag,
                                                        (short)0, acc[dt], false, false);
    }
  }

  // normalize and store to [B*S][1024] bf16 for the output projection
#pragma unroll
  for (int dt = 0; dt < 4; ++dt)
#pragma unroll
    for (int r = 0; r < 8; ++r) {
      const int s = q0 + rowM + r;
      const float ov = acc[dt][r] / l_i[r];
      ob[(size_t)(b * SEQ + s) * D_MODEL + h * HEAD_DIM + dt * 16 + n] = (bf16_t)ov;
    }
}

// out = attn @ W_out + b_out (fp32 output). 32x64 tile per wave.
__global__ __launch_bounds__(32) void out_gemm_kernel(
    const bf16_t* __restrict__ A,     // [8192][1024] bf16
    const bf16_t* __restrict__ Bt,    // [1024][1024] bf16 (W_out^T)
    const float*  __restrict__ bias,  // [1024]
    float* __restrict__ out)          // [8192][1024] fp32
{
  const int lane = threadIdx.x;
  const int mt = blockIdx.x;          // 256
  const int nt = blockIdx.y;          // 16
  v8f acc[2][4] = {};

  for (int k0 = 0; k0 < D_MODEL; k0 += 32) {
    AFrag af0 = load_a_frag(A, D_MODEL, mt * 32,      k0, lane);
    AFrag af1 = load_a_frag(A, D_MODEL, mt * 32 + 16, k0, lane);
#pragma unroll
    for (int j = 0; j < 4; ++j) {
      v16bf bfrag = load_b_frag(Bt, D_MODEL, nt * 64 + j * 16, k0, lane);
      acc[0][j] = __builtin_amdgcn_wmma_f32_16x16x32_bf16(false, af0.v, false, bfrag,
                                                          (short)0, acc[0][j], false, false);
      acc[1][j] = __builtin_amdgcn_wmma_f32_16x16x32_bf16(false, af1.v, false, bfrag,
                                                          (short)0, acc[1][j], false, false);
    }
  }

  const int n    = lane & 15;
  const int rowM = (lane >> 4) * 8;
#pragma unroll
  for (int mi = 0; mi < 2; ++mi) {
#pragma unroll
    for (int j = 0; j < 4; ++j) {
      const int colg = nt * 64 + j * 16 + n;
      const float bv = bias[colg];
#pragma unroll
      for (int r = 0; r < 8; ++r) {
        const int m = mt * 32 + mi * 16 + rowM + r;
        out[(size_t)m * D_MODEL + colg] = acc[mi][j][r] + bv;
      }
    }
  }
}

extern "C" void kernel_launch(void* const* d_in, const int* in_sizes, int n_in,
                              void* d_out, int out_size, void* d_ws, size_t ws_size,
                              hipStream_t stream) {
  (void)in_sizes; (void)n_in; (void)out_size; (void)ws_size;

  const float* x     = (const float*)d_in[0];   // [4,2048,1024]
  const float* W_qkv = (const float*)d_in[1];   // [1024,3072]
  const float* b_qkv = (const float*)d_in[2];   // [3072]
  const float* W_out = (const float*)d_in[3];   // [1024,1024]
  const float* b_out = (const float*)d_in[4];   // [1024]
  float* out = (float*)d_out;

  const size_t szXD = (size_t)NROWS * D_MODEL;  // 8388608 elements

  char* ws = (char*)d_ws;
  bf16_t* x_bf    = (bf16_t*)ws;  ws += szXD * sizeof(bf16_t);
  bf16_t* wqkvT   = (bf16_t*)ws;  ws += (size_t)3 * D_MODEL * D_MODEL * sizeof(bf16_t);
  bf16_t* woutT   = (bf16_t*)ws;  ws += (size_t)D_MODEL * D_MODEL * sizeof(bf16_t);
  bf16_t* qbuf    = (bf16_t*)ws;  ws += szXD * sizeof(bf16_t);
  bf16_t* kbuf    = (bf16_t*)ws;  ws += szXD * sizeof(bf16_t);
  bf16_t* vtbuf   = (bf16_t*)ws;  ws += szXD * sizeof(bf16_t);
  bf16_t* attn_bf = (bf16_t*)ws;  ws += szXD * sizeof(bf16_t);

  // 1) fp32 -> bf16 conversions / weight transposes
  cvt_bf16_kernel<<<(unsigned)((szXD + 255) / 256), 256, 0, stream>>>(x, x_bf, (int)szXD);
  transpose_cvt_kernel<<<(3 * D_MODEL * D_MODEL + 255) / 256, 256, 0, stream>>>(
      W_qkv, wqkvT, D_MODEL, 3 * D_MODEL);
  transpose_cvt_kernel<<<(D_MODEL * D_MODEL + 255) / 256, 256, 0, stream>>>(
      W_out, woutT, D_MODEL, D_MODEL);

  // 2) QKV projection (WMMA, 32x64 tiles)
  qkv_gemm_kernel<<<dim3(NROWS / 32, (3 * D_MODEL) / 64), 32, 0, stream>>>(
      x_bf, wqkvT, b_qkv, qbuf, kbuf, vtbuf);

  // 3) causal flash attention (WMMA + online softmax)
  attn_kernel<<<dim3(SEQ / 16, NHEAD, BATCH), 32, 0, stream>>>(qbuf, kbuf, vtbuf, attn_bf);

  // 4) output projection (WMMA, 32x64 tiles, fp32 out)
  out_gemm_kernel<<<dim3(NROWS / 32, D_MODEL / 64), 32, 0, stream>>>(
      attn_bf, woutT, b_out, out);
}
